// CustomMultiHeadAttention_6442450944545
// MI455X (gfx1250) — matrix operand order repaired and checked by reading, not compiled, not run on hardware
//
#include <hip/hip_runtime.h>

typedef __bf16 v16bf __attribute__((ext_vector_type(16)));
typedef __bf16 v8bf  __attribute__((ext_vector_type(8)));
typedef float  v8f   __attribute__((ext_vector_type(8)));

union BF16x16 { v16bf v; v8bf h[2]; };

__device__ __forceinline__ v8f wmma_bf16(const BF16x16 &a, const BF16x16 &b, v8f c) {
    return __builtin_amdgcn_wmma_f32_16x16x32_bf16(false, a.v, false, b.v,
                                                   (short)0, c, false, false);
}

// CDNA5 async global->LDS copy (16B per lane), tracked with ASYNCcnt.
__device__ __forceinline__ void async_g2l_b128(unsigned ldsOff, const void* gaddr) {
    asm volatile("global_load_async_to_lds_b128 %0, %1, off"
                 :: "v"(ldsOff), "v"(gaddr) : "memory");
}
__device__ __forceinline__ void wait_async0() {
    asm volatile("s_wait_asynccnt 0x0" ::: "memory");
}

constexpr int Bn  = 4;
constexpr int Sq  = 2048;
constexpr int Dm  = 1024;
constexpr int Hh  = 16;
constexpr int DK  = 64;
constexpr int Pp  = 1792;     // common_prefix_len (compile-time; grid dims need it on host)
constexpr int Ncand = Sq - Pp;
constexpr int LSTR = 40;      // padded LDS row stride (bf16) to spread banks

__device__ __forceinline__ float rowMax16(float x) {
#pragma unroll
    for (int off = 1; off < 16; off <<= 1) x = fmaxf(x, __shfl_xor(x, off, 32));
    return x;
}
__device__ __forceinline__ float rowSum16(float x) {
#pragma unroll
    for (int off = 1; off < 16; off <<= 1) x += __shfl_xor(x, off, 32);
    return x;
}

// ---------------------------------------------------------------------------
// Kernel 0: one-time f32 -> bf16 conversion of the four weight matrices.
// grid = (512, 4), block 256, 8 elems/thread.
// ---------------------------------------------------------------------------
__global__ __launch_bounds__(256) void wcvt_kernel(
    const float* __restrict__ w0, const float* __restrict__ w1,
    const float* __restrict__ w2, const float* __restrict__ w3,
    __bf16* __restrict__ o0, __bf16* __restrict__ o1,
    __bf16* __restrict__ o2, __bf16* __restrict__ o3)
{
    const int w = blockIdx.y;
    const float* src = w == 0 ? w0 : w == 1 ? w1 : w == 2 ? w2 : w3;
    __bf16*      dst = w == 0 ? o0 : w == 1 ? o1 : w == 2 ? o2 : o3;
    size_t i = ((size_t)blockIdx.x * 256 + threadIdx.x) * 8;
    const float* s = src + i;
    v8bf pk;
#pragma unroll
    for (int e = 0; e < 8; ++e) pk[e] = (__bf16)s[e];
    *(v8bf*)(dst + i) = pk;
}

// ---------------------------------------------------------------------------
// Kernel 1: fused QKV projection.  Y[b,s,o] = sum_d X[b,s,d]*W[o,d] + bias[o]
// Q,K -> bf16 (b,h,s,dk);  V -> bf16 TRANSPOSED (b,h,dk,s) so the PV operand
// in attention is a contiguous load.  grid = (M/64, Dm/128, 3), block 256.
// ---------------------------------------------------------------------------
__global__ __launch_bounds__(256) void qkv_proj_kernel(
    const float* __restrict__ q, const float* __restrict__ k, const float* __restrict__ v,
    const __bf16* __restrict__ Wqb, const __bf16* __restrict__ Wkb, const __bf16* __restrict__ Wvb,
    const float* __restrict__ bq, const float* __restrict__ bk, const float* __restrict__ bv,
    __bf16* __restrict__ Qb, __bf16* __restrict__ Kb, __bf16* __restrict__ Vtb)
{
    const int which = blockIdx.z;
    const float*  X    = which == 0 ? q   : (which == 1 ? k   : v);
    const __bf16* W    = which == 0 ? Wqb : (which == 1 ? Wkb : Wvb);
    const float*  bias = which == 0 ? bq  : (which == 1 ? bk  : bv);

    __shared__ __bf16 As[64  * LSTR];
    __shared__ __bf16 Bs[128 * LSTR];

    const int t    = threadIdx.x;
    const int wave = t >> 5, lane = t & 31;
    const int wm   = wave >> 2, wn = wave & 3;       // 2 x 4 wave grid
    const int hl   = lane >> 4, l16 = lane & 15;
    const int m0   = blockIdx.x * 64;
    const int n0   = blockIdx.y * 128;

    v8f acc[2][2] = {};

    for (int k0 = 0; k0 < Dm; k0 += 32) {
        // B tile 128x32 bf16: async copy straight into LDS (2 x 16B / thread)
        {
            int idx = t * 16;
            int r = idx >> 5, c = idx & 31;
            const __bf16* src = W + (size_t)(n0 + r) * Dm + k0 + c;
            async_g2l_b128((unsigned)(size_t)&Bs[r * LSTR + c],     src);
            async_g2l_b128((unsigned)(size_t)&Bs[r * LSTR + c + 8], src + 8);
        }
        // A tile 64x32: 8 f32 per thread, convert to bf16, ds_store
        {
            int idx = t * 8;
            int r = idx >> 5, c = idx & 31;
            const float* src = X + (size_t)(m0 + r) * Dm + k0 + c;
            v8bf pk;
#pragma unroll
            for (int i = 0; i < 8; ++i) pk[i] = (__bf16)src[i];
            *(v8bf*)&As[r * LSTR + c] = pk;
        }
        wait_async0();
        __syncthreads();

        BF16x16 aF[2], bF[2];
#pragma unroll
        for (int mt = 0; mt < 2; ++mt) {
            const __bf16* base = &As[(wm * 32 + mt * 16 + l16) * LSTR];
            aF[mt].h[0] = *(const v8bf*)(base + hl * 16);
            aF[mt].h[1] = *(const v8bf*)(base + hl * 16 + 8);
        }
#pragma unroll
        for (int nt = 0; nt < 2; ++nt) {
            const __bf16* base = &Bs[(wn * 32 + nt * 16 + l16) * LSTR] + hl * 16;
            bF[nt].h[0] = *(const v8bf*)(base);
            bF[nt].h[1] = *(const v8bf*)(base + 8);
        }
#pragma unroll
        for (int mt = 0; mt < 2; ++mt)
#pragma unroll
            for (int nt = 0; nt < 2; ++nt)
                acc[mt][nt] = wmma_bf16(aF[mt], bF[nt], acc[mt][nt]);
        __syncthreads();
    }

    // epilogue
#pragma unroll
    for (int mt = 0; mt < 2; ++mt) {
#pragma unroll
        for (int nt = 0; nt < 2; ++nt) {
            int o = n0 + wn * 32 + nt * 16 + l16;
            float bb = bias[o];
            int h = o >> 6, d = o & 63;
            int mbase = m0 + wm * 32 + mt * 16 + hl * 8;
            int bidx = mbase >> 11;            // / Sq
            int sbase = mbase & (Sq - 1);
            if (which == 2) {
                // V: transposed (b,h,dk,s) -> r maps to consecutive s: one 16B store
                v8bf pk;
#pragma unroll
                for (int r = 0; r < 8; ++r) pk[r] = (__bf16)(acc[mt][nt][r] + bb);
                *(v8bf*)(Vtb + ((size_t)(bidx * Hh + h) * DK + d) * Sq + sbase) = pk;
            } else {
                __bf16* Out = (which == 0) ? Qb : Kb;
#pragma unroll
                for (int r = 0; r < 8; ++r) {
                    float val = acc[mt][nt][r] + bb;
                    Out[(((size_t)(bidx * Hh + h) * Sq + (sbase + r)) << 6) + d] = (__bf16)val;
                }
            }
        }
    }
}

// ---------------------------------------------------------------------------
// Kernel 2: flash attention.  One wave per 16-query tile, 32-key tiles.
// mode 0: prefix rows [0,P) causal.  mode 1: candidate rows [P,S): prefix + diag.
// ---------------------------------------------------------------------------
__global__ __launch_bounds__(256) void attn_kernel(
    const __bf16* __restrict__ Qb, const __bf16* __restrict__ Kb,
    const __bf16* __restrict__ Vtb, __bf16* __restrict__ Ctx, int mode)
{
    __shared__ __bf16 ldsbuf[8 * 16 * LSTR];       // per-wave P-tile slices only

    const int t = threadIdx.x, wave = t >> 5, lane = t & 31;
    const int hl = lane >> 4, l16 = lane & 15;
    __bf16* Pt = &ldsbuf[wave * 16 * LSTR];

    const int tilesPerBH = (mode == 0) ? (Pp / 16) : (Ncand / 16);
    const int tileId = blockIdx.x * 8 + wave;
    const int bh = tileId / tilesPerBH;
    const int qt = tileId % tilesPerBH;
    const int q0 = (mode == 0) ? qt * 16 : (Pp + qt * 16);

    // Q fragments (A operand), loaded once
    BF16x16 qF[2];
    {
        const __bf16* qr = Qb + ((size_t)bh * Sq + q0 + l16) * DK;
#pragma unroll
        for (int ks = 0; ks < 2; ++ks) {
            qF[ks].h[0] = *(const v8bf*)(qr + ks * 32 + hl * 16);
            qF[ks].h[1] = *(const v8bf*)(qr + ks * 32 + hl * 16 + 8);
        }
    }

    float mrow[8], lrow[8];
    v8f ctx[4] = {};
#pragma unroll
    for (int r = 0; r < 8; ++r) { mrow[r] = -3.0e38f; lrow[r] = 0.f; }

    const int kend = (mode == 0) ? (q0 + 16) : Pp;
    for (int j = 0; j < kend; j += 32) {
        // prefetch next K rows / V^T rows while this tile computes
        if (j + 32 < kend) {
            __builtin_prefetch(Kb + ((size_t)bh * Sq + j + 32 + lane) * DK, 0, 1);
            __builtin_prefetch(Vtb + ((size_t)bh * DK + lane) * Sq + j + 32, 0, 1);
            __builtin_prefetch(Vtb + ((size_t)bh * DK + 32 + lane) * Sq + j + 32, 0, 1);
        }

        // ---- QK^T: B operand straight from row-major K ----
        BF16x16 kF[2][2];
#pragma unroll
        for (int nt = 0; nt < 2; ++nt) {
            const __bf16* kr = Kb + ((size_t)bh * Sq + j + nt * 16 + l16) * DK;
#pragma unroll
            for (int ks = 0; ks < 2; ++ks) {
                kF[nt][ks].h[0] = *(const v8bf*)(kr + ks * 32 + hl * 16);
                kF[nt][ks].h[1] = *(const v8bf*)(kr + ks * 32 + hl * 16 + 8);
            }
        }
        v8f s0 = {}, s1 = {};
        s0 = wmma_bf16(qF[0], kF[0][0], s0);
        s0 = wmma_bf16(qF[1], kF[0][1], s0);
        s1 = wmma_bf16(qF[0], kF[1][0], s1);
        s1 = wmma_bf16(qF[1], kF[1][1], s1);

        // ---- scale + causal mask ----
#pragma unroll
        for (int r = 0; r < 8; ++r) {
            float a = s0[r] * 0.125f, b = s1[r] * 0.125f;
            if (mode == 0) {
                int qrow = q0 + r + hl * 8;
                if (j + l16 > qrow)      a = -3.0e38f;
                if (j + 16 + l16 > qrow) b = -3.0e38f;
            }
            s0[r] = a; s1[r] = b;
        }

        // ---- online softmax ----
        float alpha[8];
#pragma unroll
        for (int r = 0; r < 8; ++r) {
            float tmax = rowMax16(fmaxf(s0[r], s1[r]));
            float mnew = fmaxf(mrow[r], tmax);
            alpha[r] = __expf(mrow[r] - mnew);
            mrow[r] = mnew;
            float p0 = __expf(s0[r] - mnew);
            float p1 = __expf(s1[r] - mnew);
            s0[r] = p0; s1[r] = p1;
            lrow[r] = lrow[r] * alpha[r] + rowSum16(p0 + p1);
        }
#pragma unroll
        for (int nt = 0; nt < 4; ++nt)
#pragma unroll
            for (int r = 0; r < 8; ++r) ctx[nt][r] *= alpha[r];

        // ---- P: C-layout -> A-layout via per-wave LDS slice ----
#pragma unroll
        for (int r = 0; r < 8; ++r) {
            int m = r + hl * 8;
            Pt[m * LSTR + l16]      = (__bf16)s0[r];
            Pt[m * LSTR + 16 + l16] = (__bf16)s1[r];
        }
        BF16x16 pF;
        {
            const __bf16* pr = Pt + l16 * LSTR;
            pF.h[0] = *(const v8bf*)(pr + hl * 16);
            pF.h[1] = *(const v8bf*)(pr + hl * 16 + 8);
        }

        // ---- ctx += P * V : B operand contiguous from transposed V ----
#pragma unroll
        for (int nt = 0; nt < 4; ++nt) {
            BF16x16 vF;
            const __bf16* vb = Vtb + ((size_t)bh * DK + nt * 16 + l16) * Sq + j + hl * 16;
            vF.h[0] = *(const v8bf*)(vb);
            vF.h[1] = *(const v8bf*)(vb + 8);
            ctx[nt] = wmma_bf16(pF, vF, ctx[nt]);
        }
    }

    // ---- candidate diagonal self-key ----
    if (mode == 1) {
        float diag[8];
#pragma unroll
        for (int r = 0; r < 8; ++r) {
            int s = q0 + r + hl * 8;
            const __bf16* qr = Qb + ((size_t)bh * Sq + s) * DK + l16 * 4;
            const __bf16* kr = Kb + ((size_t)bh * Sq + s) * DK + l16 * 4;
            float a = 0.f;
#pragma unroll
            for (int d = 0; d < 4; ++d) a += (float)qr[d] * (float)kr[d];
            diag[r] = rowSum16(a) * 0.125f;
        }
#pragma unroll
        for (int r = 0; r < 8; ++r) {
            float mnew = fmaxf(mrow[r], diag[r]);
            float al = __expf(mrow[r] - mnew);
            float w  = __expf(diag[r] - mnew);
            lrow[r] = lrow[r] * al + w;
            mrow[r] = mnew;
            int s = q0 + r + hl * 8;
#pragma unroll
            for (int nt = 0; nt < 4; ++nt) {
                float vv = (float)Vtb[((size_t)bh * DK + nt * 16 + l16) * Sq + s];
                ctx[nt][r] = ctx[nt][r] * al + w * vv;
            }
        }
    }

    // ---- normalize, store bf16 ctx in (b, s, h*dk) row-major ----
    const int bidx = bh >> 4, h = bh & 15;
#pragma unroll
    for (int r = 0; r < 8; ++r) {
        float inv = 1.0f / lrow[r];
        int s = q0 + r + hl * 8;
        __bf16* dst = Ctx + ((size_t)(bidx * Sq + s)) * Dm + h * DK;
#pragma unroll
        for (int nt = 0; nt < 4; ++nt)
            dst[nt * 16 + l16] = (__bf16)(ctx[nt][r] * inv);
    }
}

// ---------------------------------------------------------------------------
// Kernel 3: output projection.  out[m,o] = sum_d ctx[m,d]*Wo[o,d] + bo[o], f32.
// Both tiles already bf16 -> both staged with async global->LDS copies.
// ---------------------------------------------------------------------------
__global__ __launch_bounds__(256) void out_proj_kernel(
    const __bf16* __restrict__ Ctx, const __bf16* __restrict__ Wob,
    const float* __restrict__ bo, float* __restrict__ Yout)
{
    __shared__ __bf16 As[64  * LSTR];
    __shared__ __bf16 Bs[128 * LSTR];

    const int t = threadIdx.x;
    const int wave = t >> 5, lane = t & 31;
    const int wm = wave >> 2, wn = wave & 3;
    const int hl = lane >> 4, l16 = lane & 15;
    const int m0 = blockIdx.x * 64;
    const int n0 = blockIdx.y * 128;

    v8f acc[2][2] = {};

    for (int k0 = 0; k0 < Dm; k0 += 32) {
        {
            int idx = t * 8;
            int r = idx >> 5, c = idx & 31;
            async_g2l_b128((unsigned)(size_t)&As[r * LSTR + c],
                           Ctx + (size_t)(m0 + r) * Dm + k0 + c);
        }
        {
            int idx = t * 16;
            int r = idx >> 5, c = idx & 31;
            const __bf16* src = Wob + (size_t)(n0 + r) * Dm + k0 + c;
            async_g2l_b128((unsigned)(size_t)&Bs[r * LSTR + c],     src);
            async_g2l_b128((unsigned)(size_t)&Bs[r * LSTR + c + 8], src + 8);
        }
        wait_async0();
        __syncthreads();

        BF16x16 aF[2], bF[2];
#pragma unroll
        for (int mt = 0; mt < 2; ++mt) {
            const __bf16* base = &As[(wm * 32 + mt * 16 + l16) * LSTR];
            aF[mt].h[0] = *(const v8bf*)(base + hl * 16);
            aF[mt].h[1] = *(const v8bf*)(base + hl * 16 + 8);
        }
#pragma unroll
        for (int nt = 0; nt < 2; ++nt) {
            const __bf16* base = &Bs[(wn * 32 + nt * 16 + l16) * LSTR] + hl * 16;
            bF[nt].h[0] = *(const v8bf*)(base);
            bF[nt].h[1] = *(const v8bf*)(base + 8);
        }
#pragma unroll
        for (int mt = 0; mt < 2; ++mt)
#pragma unroll
            for (int nt = 0; nt < 2; ++nt)
                acc[mt][nt] = wmma_bf16(aF[mt], bF[nt], acc[mt][nt]);
        __syncthreads();
    }

#pragma unroll
    for (int mt = 0; mt < 2; ++mt) {
#pragma unroll
        for (int nt = 0; nt < 2; ++nt) {
            int o = n0 + wn * 32 + nt * 16 + l16;
            float bb = bo[o];
#pragma unroll
            for (int r = 0; r < 8; ++r) {
                int m = m0 + wm * 32 + mt * 16 + r + hl * 8;
                Yout[(size_t)m * Dm + o] = acc[mt][nt][r] + bb;
            }
        }
    }
}

// ---------------------------------------------------------------------------
extern "C" void kernel_launch(void* const* d_in, const int* in_sizes, int n_in,
                              void* d_out, int out_size, void* d_ws, size_t ws_size,
                              hipStream_t stream)
{
    (void)in_sizes; (void)n_in; (void)out_size; (void)ws_size;
    const float* q  = (const float*)d_in[0];
    const float* k  = (const float*)d_in[1];
    const float* v  = (const float*)d_in[2];
    const float* Wq = (const float*)d_in[3];
    const float* bq = (const float*)d_in[4];
    const float* Wk = (const float*)d_in[5];
    const float* bk = (const float*)d_in[6];
    const float* Wv = (const float*)d_in[7];
    const float* bv = (const float*)d_in[8];
    const float* Wo = (const float*)d_in[9];
    const float* bo = (const float*)d_in[10];
    float* out = (float*)d_out;

    const size_t headElems = (size_t)Bn * Hh * Sq * DK;   // 8,388,608
    const size_t wElems    = (size_t)Dm * Dm;             // 1,048,576
    __bf16* Qb  = (__bf16*)d_ws;
    __bf16* Kb  = Qb + headElems;
    __bf16* Vtb = Kb + headElems;          // (b,h,dk,s) transposed layout
    __bf16* Cx  = Vtb + headElems;         // (B*S, D) bf16
    __bf16* Wqb = Cx + headElems;
    __bf16* Wkb = Wqb + wElems;
    __bf16* Wvb = Wkb + wElems;
    __bf16* Wob = Wvb + wElems;

    // 0) weights -> bf16 once
    wcvt_kernel<<<dim3(512, 4, 1), 256, 0, stream>>>(Wq, Wk, Wv, Wo, Wqb, Wkb, Wvb, Wob);

    // 1) fused QKV projection (V written transposed)
    dim3 gp((Bn * Sq) / 64, Dm / 128, 3);
    qkv_proj_kernel<<<gp, 256, 0, stream>>>(q, k, v, Wqb, Wkb, Wvb, bq, bk, bv, Qb, Kb, Vtb);

    // 2) flash attention: prefix (causal) then candidates (prefix keys + diag)
    attn_kernel<<<(Bn * Hh * (Pp / 16)) / 8, 256, 0, stream>>>(Qb, Kb, Vtb, Cx, 0);
    attn_kernel<<<(Bn * Hh * (Ncand / 16)) / 8, 256, 0, stream>>>(Qb, Kb, Vtb, Cx, 1);

    // 3) output projection -> f32
    out_proj_kernel<<<dim3((Bn * Sq) / 64, Dm / 128, 1), 256, 0, stream>>>(Cx, Wob, bo, out);
}
